// Pool_52149492908808
// MI455X (gfx1250) — compile-verified, matrix operand order checked by reading, not run on hardware
//
#include <hip/hip_runtime.h>
#include <hip/hip_bf16.h>

typedef __attribute__((ext_vector_type(16))) _Float16 v16h;
typedef __attribute__((ext_vector_type(8)))  _Float16 v8h;
typedef __attribute__((ext_vector_type(8)))  float    v8f;
typedef __attribute__((ext_vector_type(4)))  unsigned int u32x4;
typedef __attribute__((ext_vector_type(8)))  int      i32x8;
typedef __attribute__((ext_vector_type(4)))  int      i32x4;

#define EPS   1e-5f
#define BATCH 8
#define HH    256
#define WWD   256
#define CIN   3
#define CMID  128

#if defined(__has_builtin)
#if __has_builtin(__builtin_amdgcn_tensor_load_to_lds) && \
    __has_builtin(__builtin_amdgcn_s_wait_tensorcnt)
#define HAVE_TDM 1
#endif
#endif

// -----------------------------------------------------------------------------
// Prep kernel (runs once, ~2K threads): pack weights into WMMA per-lane
// fragments (f16, contiguous K order: element e <-> K = (lane<16?0:16)+e) and
// fold BN into per-channel scale/shift. Removes all divergent weight gathers,
// f32->f16 converts and v_rsq from the hot conv loops.
//   wfragA: [conv][nt 0..7][lane 0..31][16]  A = weights of Conv(3->128)
//   wfragB: [kk 0..35][lane 0..31][16]       B = weights of Conv(128->3)
//   bninv/bnsh: [conv][128]                  BN scale/shift for front convs
// -----------------------------------------------------------------------------
__global__ void k_prep(
    const float* __restrict__ w1, const float* __restrict__ g1,
    const float* __restrict__ b1, const float* __restrict__ m1,
    const float* __restrict__ v1,
    const float* __restrict__ w2, const float* __restrict__ g2,
    const float* __restrict__ b2, const float* __restrict__ m2,
    const float* __restrict__ v2,
    const float* __restrict__ wp,
    _Float16* __restrict__ wfragA, _Float16* __restrict__ wfragB,
    float* __restrict__ bninv, float* __restrict__ bnsh)
{
    const int i = blockIdx.x * 256 + threadIdx.x;
    if (i < 512) {                               // front-conv A fragments
        const int lane = i & 31, nt = (i >> 5) & 7, conv = i >> 8;
        const float* wgt = conv ? w2 : w1;
        const int ml = lane & 15, kb = (lane < 16) ? 0 : 16;
        const int oa = nt * 16 + ml;             // output channel (A row M)
        v16h f;
#pragma unroll
        for (int e = 0; e < 16; ++e) {
            const int k = kb + e;                // K = c*9 + kh*3 + kw
            f[e] = (_Float16)((k < 27) ? wgt[oa * 27 + k] : 0.f);
        }
        *(v16h*)&wfragA[(size_t)i * 16] = f;
    } else if (i < 512 + 1152) {                 // post-conv B fragments
        const int j = i - 512;
        const int lane = j & 31, kk = j >> 5;    // kk = tap*4 + cq
        const int nl = lane & 15, kb = (lane < 16) ? 0 : 16;
        const int p = kk >> 2, cbase = (kk & 3) * 32;
        v16h f;
#pragma unroll
        for (int e = 0; e < 16; ++e) {
            const int c = cbase + kb + e;
            f[e] = (_Float16)((nl < 3) ? wp[nl * (CMID * 9) + c * 9 + p] : 0.f);
        }
        *(v16h*)&wfragB[(size_t)j * 16] = f;
    } else if (i < 512 + 1152 + 256) {           // folded BN for front convs
        const int j = i - (512 + 1152);          // conv*128 + oc
        const int conv = j >> 7, oc = j & 127;
        const float* gg = conv ? g2 : g1;
        const float* bb = conv ? b2 : b1;
        const float* mm = conv ? m2 : m1;
        const float* vv = conv ? v2 : v1;
        const float inv = gg[oc] * rsqrtf(vv[oc] + EPS);
        bninv[j] = inv;
        bnsh[j]  = bb[oc] - mm[oc] * inv;
    }
}

// -----------------------------------------------------------------------------
// Kernel 1: Conv(3->128, 3x3, pad 1) + BN + ReLU for both branches (WMMA).
// Block = 8 waves covering 128 w at fixed (b,h,conv). Input slab
// (3ch x 3rows x 130w) staged once in LDS as f16; per wave: one im2col B
// fragment from LDS, then 8x { load A fragment (32B), wmma, BN, 16B store }.
// Output f16 NHWC so channels are contiguous for the later WMMA consumer.
// -----------------------------------------------------------------------------
__global__ __launch_bounds__(256) void k_front(
    const float* __restrict__ x,
    const _Float16* __restrict__ wfragA,
    const float* __restrict__ bninv, const float* __restrict__ bnsh,
    _Float16* __restrict__ p1, _Float16* __restrict__ p2)
{
    __shared__ _Float16 xs[9 * 132];             // [c*3+kh][w: -1..128] (+pad)

    const int lane  = threadIdx.x;
    const int wv    = threadIdx.y;               // wave id 0..7
    const int h     = blockIdx.y;
    const int b     = blockIdx.z & 7;
    const int conv  = blockIdx.z >> 3;
    const int wbase = blockIdx.x * 128;

    // stage x rows h-1..h+1, cols wbase-1..wbase+128, 3 channels (f32 -> f16)
    const int t = wv * 32 + lane;
    for (int i = t; i < 9 * 130; i += 256) {
        const int cr = i / 130;                  // c*3 + kh
        const int lw = i - cr * 130;             // 0..129
        const int c  = cr / 3, kh = cr - c * 3;
        const int hh = h + kh - 1;
        const int wg = wbase - 1 + lw;
        float val = 0.f;
        if (hh >= 0 && hh < HH && wg >= 0 && wg < WWD)
            val = x[((b * CIN + c) * HH + hh) * WWD + wg];
        xs[cr * 132 + lw] = (_Float16)val;
    }
    __syncthreads();

    const int ml = lane & 15;                    // pixel index (B: N)
    const int kb = (lane < 16) ? 0 : 16;         // K half of this lane
    const int wloc = wv * 16;

    // B = im2col fragment from LDS
    v16h Bm;
#pragma unroll
    for (int e = 0; e < 16; ++e) {
        const int k = kb + e;
        _Float16 val = (_Float16)0.f;
        if (k < 27) {
            const int c  = k / 9, r = k - c * 9;
            const int kh = r / 3, kw = r - kh * 3;
            val = xs[(c * 3 + kh) * 132 + (wloc + ml + kw)];
        }
        Bm[e] = val;
    }

    _Float16* p = conv ? p2 : p1;
    const size_t obase = ((size_t)(b * HH + h) * WWD + (wbase + wloc + ml)) * CMID;

#pragma unroll 1
    for (int nt = 0; nt < 8; ++nt) {             // 8 tiles of 16 out-channels
        const v16h Am = *(const v16h*)&wfragA[(size_t)(((conv << 3) + nt) * 32 + lane) * 16];
        v8f acc = {};
        acc = __builtin_amdgcn_wmma_f32_16x16x32_f16(
            false, Am, false, Bm, (short)0, acc, false, false);

        // D: this lane = pixel (wbase+wloc+ml), channels ob..ob+7
        const int ob = nt * 16 + ((lane < 16) ? 0 : 8);
        const v8f inv = *(const v8f*)&bninv[(conv << 7) + ob];
        const v8f sh  = *(const v8f*)&bnsh[(conv << 7) + ob];
        v8h outp;
#pragma unroll
        for (int g = 0; g < 8; ++g) {
            const float val = acc[g] * inv[g] + sh[g];
            outp[g] = (_Float16)(val > 0.f ? val : 0.f);
        }
        *(v8h*)&p[obase + ob] = outp;
    }
}

// -----------------------------------------------------------------------------
// Kernel 2: TopPool = reverse cummax over H, in place on p1 (f16 NHWC).
// -----------------------------------------------------------------------------
__global__ void k_toppool(_Float16* __restrict__ p1)
{
    const int idx = blockIdx.x * blockDim.x + threadIdx.x;  // 8*256*16
    const int c8 = idx & 15;
    const int w  = (idx >> 4) & 255;
    const int b  = idx >> 12;

    v8h r = *(v8h*)&p1[(((size_t)b * HH + 255) * WWD + w) * CMID + c8 * 8];
    for (int h = 254; h >= 0; --h) {
        const size_t a = (((size_t)b * HH + h) * WWD + w) * CMID + c8 * 8;
        v8h v = *(v8h*)&p1[a];
#pragma unroll
        for (int j = 0; j < 8; ++j) r[j] = (v[j] > r[j]) ? v[j] : r[j];
        *(v8h*)&p1[a] = r;
    }
}

// -----------------------------------------------------------------------------
// Kernel 3: LeftPool (reverse cummax over W) on p2 fused with s = p1'+p2'
// written into p1.
// -----------------------------------------------------------------------------
__global__ void k_leftpool_add(const _Float16* __restrict__ p2,
                               _Float16* __restrict__ p1)
{
    const int idx = blockIdx.x * blockDim.x + threadIdx.x;  // 8*256*16
    const int c8 = idx & 15;
    const int h  = (idx >> 4) & 255;
    const int b  = idx >> 12;

    v8h r;
    {
        const size_t a = (((size_t)b * HH + h) * WWD + 255) * CMID + c8 * 8;
        r = *(const v8h*)&p2[a];
        v8h sv = *(v8h*)&p1[a];
#pragma unroll
        for (int j = 0; j < 8; ++j) sv[j] = (_Float16)((float)sv[j] + (float)r[j]);
        *(v8h*)&p1[a] = sv;
    }
    for (int w = 254; w >= 0; --w) {
        const size_t a = (((size_t)b * HH + h) * WWD + w) * CMID + c8 * 8;
        v8h v  = *(const v8h*)&p2[a];
        v8h sv = *(v8h*)&p1[a];
#pragma unroll
        for (int j = 0; j < 8; ++j) {
            r[j]  = (v[j] > r[j]) ? v[j] : r[j];
            sv[j] = (_Float16)((float)sv[j] + (float)r[j]);
        }
        *(v8h*)&p1[a] = sv;
    }
}

// -----------------------------------------------------------------------------
// Kernel 4: Conv(128->3, 3x3)+BN + skip 1x1 Conv(3->3)+BN, ReLU -> y (f32 NHWC3).
// 8h x 16w tile per block. The s halo tile (10 rows x 2304 contiguous f16,
// 46 KB) is fetched by the Tensor Data Mover for interior blocks (one
// tensor_load_to_lds + s_wait_tensorcnt, issued by wave 0), manual
// zero-padded loads at the image border. 36 WMMAs per wave, B fragments
// prepacked in workspace (L2-resident).
// -----------------------------------------------------------------------------
__global__ __launch_bounds__(256) void k_post(
    const _Float16* __restrict__ s,              // p1 buffer: p1'+p2', NHWC
    const float* __restrict__ x,
    const _Float16* __restrict__ wfragB,
    const float* __restrict__ gp, const float* __restrict__ bp,
    const float* __restrict__ mp, const float* __restrict__ vp,
    const float* __restrict__ wsk, const float* __restrict__ gs,
    const float* __restrict__ bs, const float* __restrict__ ms,
    const float* __restrict__ vs,
    float* __restrict__ y)
{
    __shared__ _Float16 tile[10 * 18 * CMID];    // 46080 B, LDS offset 0

    const int t  = threadIdx.x;
    const int w0 = blockIdx.x * 16;
    const int h0 = blockIdx.y * 8;
    const int b  = blockIdx.z;

    const bool interior = (blockIdx.x >= 1) & (blockIdx.x <= 14) &
                          (blockIdx.y >= 1) & (blockIdx.y <= 30);
#ifdef HAVE_TDM
    if (interior) {
        if (t < 32) {                            // wave 0 issues the DMA
            // D# per ISA 8.3/8.4: 2D tile, rows of 18*128 f16 are contiguous
            // in NHWC; 10 rows with stride W*128 elements, data_size = 2B.
            const unsigned long long ga = (unsigned long long)(uintptr_t)
                (s + (((size_t)b * HH + (h0 - 1)) * WWD + (w0 - 1)) * CMID);
            const u32x4 g0 = {
                1u,                                            // count=1
                0u,                                            // lds_addr
                (unsigned)(ga & 0xFFFFFFFFu),                  // global_addr lo
                (unsigned)((ga >> 32) & 0x1FFFFFFu) | (2u << 30) }; // hi|type=2
            const i32x8 g1 = {
                (int)0x00010000,        // data_size = 1 (2 bytes)
                (int)0x80000000u,       // tensor_dim0 = 32768 (lo16 << 16)
                (int)(HH << 16),        // tensor_dim1 = 256 (lo16 << 16)
                (int)(2304 << 16),      // tile_dim0 = 18*128 (<< 16)
                10,                     // tile_dim1 = 10 rows
                WWD * CMID,             // tensor_dim0_stride = 32768
                0, 0 };
            const i32x4 gz4 = {0, 0, 0, 0};
            const i32x8 gz8 = {0, 0, 0, 0, 0, 0, 0, 0};
            __builtin_amdgcn_tensor_load_to_lds(g0, g1, gz4, gz4, gz8, 0);
            __builtin_amdgcn_s_wait_tensorcnt(0);
        }
        __syncthreads();
    } else
#endif
    {
        // manual halo load with zero padding: 180 positions x 16 8-half chunks
        for (int i = t; i < 180 * 16; i += 256) {
            const int pos = i >> 4, c16 = i & 15;
            const int pr = pos / 18, pc = pos - pr * 18;
            const int hh = h0 - 1 + pr, ww = w0 - 1 + pc;
            v8h v;
            if (hh >= 0 && hh < HH && ww >= 0 && ww < WWD)
                v = *(const v8h*)&s[(((size_t)b * HH + hh) * WWD + ww) * CMID + c16 * 8];
            else
#pragma unroll
                for (int j = 0; j < 8; ++j) v[j] = (_Float16)0.f;
            *(v8h*)&tile[pos * CMID + c16 * 8] = v;
        }
        __syncthreads();
    }

    const int lane = t & 31;
    const int hr   = t >> 5;                     // wave id -> h row
    const int nl   = lane & 15;                  // A: pixel; B/D: out channel
    const int kb   = (lane < 16) ? 0 : 16;
    const int h    = h0 + hr;

    v8f acc = {};
#pragma unroll 1
    for (int p9 = 0; p9 < 9; ++p9) {             // (kh,kw) tap
        const int kh = p9 / 3, kw = p9 - kh * 3;
        const _Float16* arow = &tile[((hr + kh) * 18 + (nl + kw)) * CMID];
#pragma unroll
        for (int q = 0; q < 4; ++q) {            // 4 chunks of 32 channels
            const v16h Am = *(const v16h*)&arow[q * 32 + kb];
            const v16h Bm = *(const v16h*)&wfragB[(size_t)(((p9 << 2) + q) * 32 + lane) * 16];
            acc = __builtin_amdgcn_wmma_f32_16x16x32_f16(
                false, Am, false, Bm, (short)0, acc, false, false);
        }
    }

    // epilogue: BN(pooled) + BN(1x1 skip on x) + ReLU -> y NHWC (3ch f32)
    if (nl < 3) {
        const float invp = gp[nl] * rsqrtf(vp[nl] + EPS);
        const float shp  = bp[nl] - mp[nl] * invp;
        const float invs = gs[nl] * rsqrtf(vs[nl] + EPS);
        const float shs  = bs[nl] - ms[nl] * invs;
        const float wk0 = wsk[nl * 3 + 0], wk1 = wsk[nl * 3 + 1], wk2 = wsk[nl * 3 + 2];
        const int mb = (lane < 16) ? 0 : 8;
#pragma unroll
        for (int g = 0; g < 8; ++g) {
            const int w = w0 + mb + g;
            const float sk = wk0 * x[((b * 3 + 0) * HH + h) * WWD + w]
                           + wk1 * x[((b * 3 + 1) * HH + h) * WWD + w]
                           + wk2 * x[((b * 3 + 2) * HH + h) * WWD + w];
            float val = acc[g] * invp + shp + sk * invs + shs;
            val = val > 0.f ? val : 0.f;
            y[(((size_t)b * HH + h) * WWD + w) * 3 + nl] = val;
        }
    }
}

// -----------------------------------------------------------------------------
// Kernel 5: final Conv(3->3, 3x3)+BN+ReLU, y (NHWC3 f32) -> out (NCHW f32).
// -----------------------------------------------------------------------------
__global__ void k_final(const float* __restrict__ y,
                        const float* __restrict__ wf, const float* __restrict__ gf,
                        const float* __restrict__ bf, const float* __restrict__ mf,
                        const float* __restrict__ vf,
                        float* __restrict__ out)
{
    const int idx = blockIdx.x * 256 + threadIdx.x;
    const int w = idx & 255, h = (idx >> 8) & 255, b = idx >> 16;

    float a0 = 0.f, a1 = 0.f, a2 = 0.f;
#pragma unroll
    for (int kh = 0; kh < 3; ++kh) {
        const int hh = h + kh - 1;
        if (hh < 0 || hh >= HH) continue;
#pragma unroll
        for (int kw = 0; kw < 3; ++kw) {
            const int ww = w + kw - 1;
            if (ww < 0 || ww >= WWD) continue;
            const float* yp = &y[(((size_t)b * HH + hh) * WWD + ww) * 3];
            const float y0 = yp[0], y1 = yp[1], y2 = yp[2];
            const int kidx = kh * 3 + kw;
            a0 += wf[0 * 27 + kidx] * y0 + wf[0 * 27 + 9 + kidx] * y1 + wf[0 * 27 + 18 + kidx] * y2;
            a1 += wf[1 * 27 + kidx] * y0 + wf[1 * 27 + 9 + kidx] * y1 + wf[1 * 27 + 18 + kidx] * y2;
            a2 += wf[2 * 27 + kidx] * y0 + wf[2 * 27 + 9 + kidx] * y1 + wf[2 * 27 + 18 + kidx] * y2;
        }
    }
    const float a[3] = {a0, a1, a2};
#pragma unroll
    for (int o = 0; o < 3; ++o) {
        const float inv = gf[o] * rsqrtf(vf[o] + EPS);
        const float val = a[o] * inv + (bf[o] - mf[o] * inv);
        out[((size_t)(b * 3 + o) * HH + h) * WWD + w] = val > 0.f ? val : 0.f;
    }
}

extern "C" void kernel_launch(void* const* d_in, const int* in_sizes, int n_in,
                              void* d_out, int out_size, void* d_ws, size_t ws_size,
                              hipStream_t stream)
{
    const float* x  = (const float*)d_in[0];
    const float* w1 = (const float*)d_in[1];
    const float* g1 = (const float*)d_in[2];
    const float* b1 = (const float*)d_in[3];
    const float* m1 = (const float*)d_in[4];
    const float* v1 = (const float*)d_in[5];
    const float* w2 = (const float*)d_in[6];
    const float* g2 = (const float*)d_in[7];
    const float* b2 = (const float*)d_in[8];
    const float* m2 = (const float*)d_in[9];
    const float* v2 = (const float*)d_in[10];
    const float* wp = (const float*)d_in[11];
    const float* gp = (const float*)d_in[12];
    const float* bp = (const float*)d_in[13];
    const float* mp = (const float*)d_in[14];
    const float* vp = (const float*)d_in[15];
    const float* wsk = (const float*)d_in[16];
    const float* gs = (const float*)d_in[17];
    const float* bs = (const float*)d_in[18];
    const float* ms = (const float*)d_in[19];
    const float* vs = (const float*)d_in[20];
    const float* wf = (const float*)d_in[21];
    const float* gf = (const float*)d_in[22];
    const float* bf = (const float*)d_in[23];
    const float* mf = (const float*)d_in[24];
    const float* vf = (const float*)d_in[25];

    const size_t planeElems = (size_t)BATCH * HH * WWD * CMID;  // 67,108,864 halves
    _Float16* p1 = (_Float16*)d_ws;
    _Float16* p2 = p1 + planeElems;
    float*    y  = (float*)(p2 + planeElems);                   // 6.3 MB f32 NHWC3
    _Float16* wfragA = (_Float16*)(y + (size_t)BATCH * HH * WWD * 3);
    _Float16* wfragB = wfragA + 512 * 16;
    float*    bninv  = (float*)(wfragB + 1152 * 16);
    float*    bnsh   = bninv + 256;

    // weight fragment / BN-fold prep (tiny)
    k_prep<<<8, 256, 0, stream>>>(w1, g1, b1, m1, v1, w2, g2, b2, m2, v2, wp,
                                  wfragA, wfragB, bninv, bnsh);
    // branch convs + BN + ReLU (WMMA)
    k_front<<<dim3(2, HH, 16), dim3(32, 8), 0, stream>>>(
        x, wfragA, bninv, bnsh, p1, p2);
    // TopPool on p1
    k_toppool<<<128, 256, 0, stream>>>(p1);
    // LeftPool on p2 fused with s = p1'+p2' (into p1)
    k_leftpool_add<<<128, 256, 0, stream>>>(p2, p1);
    // post conv + BN + skip + ReLU (WMMA + TDM tile load)
    k_post<<<dim3(WWD / 16, HH / 8, BATCH), 256, 0, stream>>>(
        p1, x, wfragB, gp, bp, mp, vp, wsk, gs, bs, ms, vs, y);
    // final conv + BN + ReLU
    k_final<<<(BATCH * HH * WWD) / 256, 256, 0, stream>>>(
        y, wf, gf, bf, mf, vf, (float*)d_out);
}
